// CNP_MLP_Mean_70342974374071
// MI455X (gfx1250) — compile-verified, matrix operand order checked by reading, not run on hardware
//
#include <hip/hip_runtime.h>
#include <stdint.h>

#define BB 32
#define LL 4096
#define UU 64
#define HXD 256
#define XDD 128
#define RDD 128
#define CC 256

typedef __attribute__((ext_vector_type(16))) __bf16 v16bf;
typedef __attribute__((ext_vector_type(2)))  __bf16 v2bf;
typedef __attribute__((ext_vector_type(8)))  float  v8f;

struct Frag32B { uint4 lo, hi; };

// Pack two f32 into packed bf16 pair (low = a, high = b).
static __device__ __forceinline__ unsigned packbf(float a, float b) {
#if __has_builtin(__builtin_amdgcn_cvt_pk_bf16_f32)
  v2bf p = __builtin_amdgcn_cvt_pk_bf16_f32(a, b);
  return __builtin_bit_cast(unsigned, p);
#else
  unsigned ua = __builtin_bit_cast(unsigned, a);
  unsigned ub = __builtin_bit_cast(unsigned, b);
  return (ua >> 16) | (ub & 0xffff0000u);   // truncate: single v_perm_b32
#endif
}

static __device__ __forceinline__ v16bf ld_frag(const uint4* p) {
  Frag32B f; f.lo = p[0]; f.hi = p[1];
  return __builtin_bit_cast(v16bf, f);
}

static __device__ __forceinline__ v16bf pack16(const float* x) {
  Frag32B f;
  unsigned* u = (unsigned*)&f;
#pragma unroll
  for (int i = 0; i < 8; ++i) u[i] = packbf(x[2 * i], x[2 * i + 1]);
  return __builtin_bit_cast(v16bf, f);
}

static __device__ __forceinline__ v16bf pack2x8(v8f a, v8f b) {
  Frag32B f;
  unsigned* u = (unsigned*)&f;
#pragma unroll
  for (int i = 0; i < 4; ++i) {
    u[i]     = packbf(a[2 * i], a[2 * i + 1]);
    u[4 + i] = packbf(b[2 * i], b[2 * i + 1]);
  }
  return __builtin_bit_cast(v16bf, f);
}

static __device__ __forceinline__ v8f wmma_bf16(v16bf a, v16bf b, v8f c) {
  return __builtin_amdgcn_wmma_f32_16x16x32_bf16(false, a, false, b, (short)0, c,
                                                 false, false);
}

// Pack row-major f32 weight [K][N] into bf16 fragments (A-frag of W^T).
// frag = kk*nNT + nt ; lane&15 = n_local ; element e: k = kk*32+(e&8?16:0)+(lane>>4)*8+(e&7)
static __device__ void stage_frags(const float* __restrict__ W, int Ndim,
                                   uint4* dst, int nKK, int nNT, int tid, int nth) {
  unsigned* d = (unsigned*)dst;
  int total = nKK * nNT * 32 * 8;
  for (int f = tid; f < total; f += nth) {
    int u    = f & 7;
    int lane = (f >> 3) & 31;
    int frag = f >> 8;
    int nt = frag % nNT, kk = frag / nNT;
    int n   = nt * 16 + (lane & 15);
    int e0  = 2 * u;
    int k0  = kk * 32 + ((e0 & 8) ? 16 : 0) + (lane >> 4) * 8 + (e0 & 7);
    d[f] = packbf(W[k0 * Ndim + n], W[(k0 + 1) * Ndim + n]);
  }
}

// W6 [128][2] -> zero-padded W6^T A-frags (M=t padded to 16, K=128 -> 4 frags)
static __device__ void stage_w6T(const float* __restrict__ W6, uint4* dst,
                                 int tid, int nth) {
  unsigned* d = (unsigned*)dst;
  for (int f = tid; f < 4 * 32 * 8; f += nth) {
    int u = f & 7, lane = (f >> 3) & 31, kk = f >> 8;
    int t  = lane & 15;
    int e0 = 2 * u;
    int k0 = kk * 32 + ((e0 & 8) ? 16 : 0) + (lane >> 4) * 8 + (e0 & 7);
    d[f] = (t < 2) ? packbf(W6[k0 * 2 + t], W6[(k0 + 1) * 2 + t]) : 0u;
  }
}

// Transposed xenc for one 16-row tile. All GEMMs computed as W^T(A) x data(B),
// so the result xT lands with lane = row (m=lane&15), element = x-column:
//   accXT[ntx][v] = x[m][ntx*16 + (lane>>4)*8 + v]   (pre-b2)
static __device__ __forceinline__ void xenc_tile_T(
    const float* __restrict__ rowfeat,   // feature row of m = lane&15
    const float* __restrict__ posrow,    // (pos+b1) row of m = lane&15
    const uint4* sW1, const uint4* sW2, int lane, v8f accXT[8]) {
  const int hl8 = (lane >> 4) * 8;

  // feature B-frags, K=64 (identical packing to an A-frag of the same row)
  v16bf af[2];
#pragma unroll
  for (int kk = 0; kk < 2; ++kk) {
    const float* p0 = rowfeat + kk * 32 + hl8;
    float4 a = *(const float4*)(p0);
    float4 b = *(const float4*)(p0 + 4);
    float4 c = *(const float4*)(p0 + 16);
    float4 d = *(const float4*)(p0 + 20);
    float tmp[16] = {a.x, a.y, a.z, a.w, b.x, b.y, b.z, b.w,
                     c.x, c.y, c.z, c.w, d.x, d.y, d.z, d.w};
    af[kk] = pack16(tmp);
  }

  v8f zero = {};
#pragma unroll
  for (int i = 0; i < 8; ++i) accXT[i] = zero;

  for (int jp = 0; jp < 8; ++jp) {       // 32 hidden columns per pass
    v8f h0 = zero, h1 = zero;
#pragma unroll
    for (int kk = 0; kk < 2; ++kk) {
      h0 = wmma_bf16(ld_frag(&sW1[((kk * 16 + 2 * jp)     * 32 + lane) * 2]), af[kk], h0);
      h1 = wmma_bf16(ld_frag(&sW1[((kk * 16 + 2 * jp + 1) * 32 + lane) * 2]), af[kk], h1);
    }
    // h^T C-layout: this lane owns h[m][jp*32 + hl8 + v] (h0) and +16 (h1)
    const float* pp = posrow + jp * 32 + hl8;
    float4 q0 = *(const float4*)(pp);
    float4 q1 = *(const float4*)(pp + 4);
    float4 q2 = *(const float4*)(pp + 16);
    float4 q3 = *(const float4*)(pp + 20);
    float hv[16] = {h0[0] + q0.x, h0[1] + q0.y, h0[2] + q0.z, h0[3] + q0.w,
                    h0[4] + q1.x, h0[5] + q1.y, h0[6] + q1.z, h0[7] + q1.w,
                    h1[0] + q2.x, h1[1] + q2.y, h1[2] + q2.z, h1[3] + q2.w,
                    h1[4] + q3.x, h1[5] + q3.y, h1[6] + q3.z, h1[7] + q3.w};
#pragma unroll
    for (int i = 0; i < 16; ++i) hv[i] = fmaxf(hv[i], 0.0f);
    v16bf hB = pack16(hv);               // B-frag of h for the next GEMM
#pragma unroll
    for (int ntx = 0; ntx < 8; ++ntx)
      accXT[ntx] = wmma_bf16(ld_frag(&sW2[((jp * 8 + ntx) * 32 + lane) * 2]), hB,
                             accXT[ntx]);
  }
}

// ---------------- pos table (b1 folded in) ----------------
__global__ void cnp_pos_kernel(const float* __restrict__ b1, float* __restrict__ posb) {
  int i = blockIdx.x * blockDim.x + threadIdx.x;
  if (i >= LL * HXD) return;
  int k = i >> 8, c = i & 255;
  float ang = (float)k * expf(-9.210340372f / (float)HXD * (float)(c & ~1));
  posb[i] = ((c & 1) ? cosf(ang) : sinf(ang)) + b1[c];
}

// ---------------- context xenc (gathered rows only) ----------------
__global__ __launch_bounds__(256) void cnp_ctx_xenc(
    const float* __restrict__ feats, const float* __restrict__ posb,
    const int* __restrict__ indexes, const float* __restrict__ W1,
    const float* __restrict__ W2, const float* __restrict__ b2,
    float* __restrict__ xctx) {
  __shared__ uint4 sW1[2 * 16 * 32 * 2];            // 32 KB
  __shared__ uint4 sW2[8 * 8 * 32 * 2];             // 64 KB
  int tid = threadIdx.x;
  stage_frags(W1, HXD, sW1, 2, 16, tid, 256);
  stage_frags(W2, XDD, sW2, 8, 8, tid, 256);
  __syncthreads();

  int wave = tid >> 5, lane = tid & 31;
  int m = lane & 15, hl8 = (lane >> 4) * 8;
  int tile = blockIdx.x * 8 + wave;                 // 512 tiles exactly
  int b = tile >> 4, tc = tile & 15;

  int myl = indexes[b * CC + tc * 16 + m];
  const float* rowfeat = feats + ((size_t)b * LL + myl) * UU;
  const float* posrow  = posb + (size_t)myl * HXD;

  v8f accXT[8];
  xenc_tile_T(rowfeat, posrow, sW1, sW2, lane, accXT);

  // this lane owns row c = tc*16+m, columns ntx*16+hl8+(0..7): vectorized stores
  int c = tc * 16 + m;
  float* orow = xctx + ((size_t)b * CC + c) * XDD;
#pragma unroll
  for (int ntx = 0; ntx < 8; ++ntx) {
    const float* bp = b2 + ntx * 16 + hl8;
    float4 u0 = *(const float4*)(bp);
    float4 u1 = *(const float4*)(bp + 4);
    v8f xx = accXT[ntx];
    float4 s0 = {xx[0] + u0.x, xx[1] + u0.y, xx[2] + u0.z, xx[3] + u0.w};
    float4 s1 = {xx[4] + u1.x, xx[5] + u1.y, xx[6] + u1.z, xx[7] + u1.w};
    *(float4*)(orow + ntx * 16 + hl8)     = s0;
    *(float4*)(orow + ntx * 16 + hl8 + 4) = s1;
  }
}

// ---------------- xyenc stage 1: r1 = relu(xy @ W3 + b3) ----------------
__global__ __launch_bounds__(128) void cnp_xyenc1(
    const float* __restrict__ xctx, const float* __restrict__ context,
    const float* __restrict__ noise, const float* __restrict__ W3,
    const float* __restrict__ b3, float* __restrict__ r1) {
  __shared__ float xy[XDD + 1];
  int bc = blockIdx.x;                              // b*CC + c
  int j = threadIdx.x;
  xy[j] = xctx[(size_t)bc * XDD + j];
  if (j == 0) xy[XDD] = context[bc] + 0.1f * noise[bc];
  __syncthreads();
  float acc = b3[j];
  for (int f = 0; f < XDD + 1; ++f) acc = fmaf(xy[f], W3[f * RDD + j], acc);
  r1[(size_t)bc * RDD + j] = fmaxf(acc, 0.0f);
}

// -------- reduce: rb = ((mean_c r1)@W4 + b4)@W5b + b5 + b2@W5a --------
__global__ __launch_bounds__(128) void cnp_reduce_r(
    const float* __restrict__ r1, const float* __restrict__ W4,
    const float* __restrict__ b4, const float* __restrict__ W5,
    const float* __restrict__ b5, const float* __restrict__ b2,
    float* __restrict__ rb) {
  __shared__ float sbuf[RDD];
  __shared__ float rbuf[RDD];
  int b = blockIdx.x, j = threadIdx.x;
  float s = 0.0f;
  for (int c = 0; c < CC; ++c) s += r1[((size_t)b * CC + c) * RDD + j];
  sbuf[j] = s * (1.0f / CC);
  __syncthreads();
  float r = b4[j];
  for (int k = 0; k < RDD; ++k) r = fmaf(sbuf[k], W4[k * RDD + j], r);
  rbuf[j] = r;
  __syncthreads();
  float v = b5[j];
  for (int k = 0; k < RDD; ++k) v = fmaf(rbuf[k], W5[(XDD + k) * RDD + j], v);
  for (int k = 0; k < XDD; ++k) v = fmaf(b2[k], W5[k * RDD + j], v);  // fold b2@W5a
  rb[b * RDD + j] = v;
}

// ---------------- fused xenc + decoder over all B*L rows ----------------
__global__ __launch_bounds__(256) void cnp_fused_dec(
    const float* __restrict__ feats, const float* __restrict__ posb,
    const float* __restrict__ W1, const float* __restrict__ W2,
    const float* __restrict__ W5, const float* __restrict__ rb,
    const float* __restrict__ W6, const float* __restrict__ b6,
    float* __restrict__ out) {
  __shared__ uint4 sW1[2 * 16 * 32 * 2];            // 32 KB
  __shared__ uint4 sW2[8 * 8 * 32 * 2];             // 64 KB
  __shared__ uint4 sW5[4 * 8 * 32 * 2];             // 32 KB (W5a rows 0..127)
  __shared__ uint4 sW6[4 * 32 * 2];                 // 4 KB  (padded W6^T)
  __shared__ float srb[BB * RDD];                   // 16 KB
  int tid = threadIdx.x;
  stage_frags(W1, HXD, sW1, 2, 16, tid, 256);
  stage_frags(W2, XDD, sW2, 8, 8, tid, 256);
  stage_frags(W5, RDD, sW5, 4, 8, tid, 256);
  stage_w6T(W6, sW6, tid, 256);
  for (int i = tid; i < BB * RDD; i += 256) srb[i] = rb[i];
  __syncthreads();

  int wave = tid >> 5, lane = tid & 31;
  int m = lane & 15, hl8 = (lane >> 4) * 8;
  float b60 = b6[0], b61 = b6[1];
  v8f zero = {};

  const int nTiles = BB * LL / 16;                  // 8192
  const int wavesTotal = gridDim.x * 8;
  for (int tile = blockIdx.x * 8 + wave; tile < nTiles; tile += wavesTotal) {
    int R = tile * 16;
    int b = R >> 12;                                // / LL
    int lpos = (R & (LL - 1)) + m;
    const float* rowfeat = feats + (size_t)(R + m) * UU;
    const float* posrow  = posb + (size_t)lpos * HXD;

    // prefetch next tile's rows while this tile computes
    {
      int nt2 = tile + wavesTotal;
      if (nt2 < nTiles) {
        __builtin_prefetch(feats + (size_t)(nt2 * 16 + m) * UU, 0, 1);
        __builtin_prefetch(posb + (size_t)((nt2 * 16 & (LL - 1)) + m) * HXD, 0, 1);
      }
    }

    v8f accXT[8];
    xenc_tile_T(rowfeat, posrow, sW1, sW2, lane, accXT);

    // decoder GEMM: dT = W5a^T x  (b2 contribution folded into rb)
    v8f accDT[8];
#pragma unroll
    for (int i = 0; i < 8; ++i) accDT[i] = zero;
#pragma unroll
    for (int kk = 0; kk < 4; ++kk) {
      v16bf xB = pack2x8(accXT[2 * kk], accXT[2 * kk + 1]);
#pragma unroll
      for (int ntd = 0; ntd < 8; ++ntd)
        accDT[ntd] = wmma_bf16(ld_frag(&sW5[((kk * 8 + ntd) * 32 + lane) * 2]), xB,
                               accDT[ntd]);
    }

    // relu(d + rb) then 4-WMMA projection against padded W6^T
    const float* rbb = srb + b * RDD;
    v8f o = zero;
#pragma unroll
    for (int kk = 0; kk < 4; ++kk) {
      float dv[16];
#pragma unroll
      for (int half = 0; half < 2; ++half) {
        int ntd = 2 * kk + half;
        const float* rp = rbb + ntd * 16 + hl8;
        float4 r0 = *(const float4*)(rp);
        float4 r1v = *(const float4*)(rp + 4);
        v8f dd = accDT[ntd];
        dv[half * 8 + 0] = fmaxf(dd[0] + r0.x, 0.0f);
        dv[half * 8 + 1] = fmaxf(dd[1] + r0.y, 0.0f);
        dv[half * 8 + 2] = fmaxf(dd[2] + r0.z, 0.0f);
        dv[half * 8 + 3] = fmaxf(dd[3] + r0.w, 0.0f);
        dv[half * 8 + 4] = fmaxf(dd[4] + r1v.x, 0.0f);
        dv[half * 8 + 5] = fmaxf(dd[5] + r1v.y, 0.0f);
        dv[half * 8 + 6] = fmaxf(dd[6] + r1v.z, 0.0f);
        dv[half * 8 + 7] = fmaxf(dd[7] + r1v.w, 0.0f);
      }
      o = wmma_bf16(ld_frag(&sW6[(kk * 32 + lane) * 2]), pack16(dv), o);
    }

    // o C-layout: lanes 0..15 hold (t=0 -> o[0], t=1 -> o[1]) for row m=lane
    if (lane < 16) {
      int row = R + lane;
      float y = o[0] + b60;
      float raw = o[1] + b61;
      float sp = (raw > 20.0f) ? raw : log1pf(expf(raw));
      out[row] = y;
      out[BB * LL + row] = 0.1f + 0.9f * sp;
    }
  }
}

extern "C" void kernel_launch(void* const* d_in, const int* in_sizes, int n_in,
                              void* d_out, int out_size, void* d_ws, size_t ws_size,
                              hipStream_t stream) {
  (void)in_sizes; (void)n_in; (void)out_size; (void)ws_size;
  const float* feats   = (const float*)d_in[0];
  const int*   indexes = (const int*)d_in[1];
  const float* context = (const float*)d_in[2];
  // d_in[3] = lens (unused)
  const float* noise   = (const float*)d_in[4];
  const float* W1 = (const float*)d_in[5];  const float* b1 = (const float*)d_in[6];
  const float* W2 = (const float*)d_in[7];  const float* b2 = (const float*)d_in[8];
  const float* W3 = (const float*)d_in[9];  const float* b3 = (const float*)d_in[10];
  const float* W4 = (const float*)d_in[11]; const float* b4 = (const float*)d_in[12];
  const float* W5 = (const float*)d_in[13]; const float* b5 = (const float*)d_in[14];
  const float* W6 = (const float*)d_in[15]; const float* b6 = (const float*)d_in[16];
  float* out = (float*)d_out;

  char* ws = (char*)d_ws;
  float* posb = (float*)ws;                                        // 4 MB
  float* xctx = (float*)(ws + (size_t)LL * HXD * 4);               // 4 MB
  float* r1   = (float*)(ws + (size_t)LL * HXD * 4
                            + (size_t)BB * CC * XDD * 4);          // 4 MB
  float* rb   = (float*)(ws + (size_t)LL * HXD * 4
                            + (size_t)BB * CC * XDD * 4
                            + (size_t)BB * CC * RDD * 4);          // 16 KB

  cnp_pos_kernel<<<(LL * HXD) / 256, 256, 0, stream>>>(b1, posb);
  cnp_ctx_xenc<<<64, 256, 0, stream>>>(feats, posb, indexes, W1, W2, b2, xctx);
  cnp_xyenc1<<<BB * CC, 128, 0, stream>>>(xctx, context, noise, W3, b3, r1);
  cnp_reduce_r<<<BB, 128, 0, stream>>>(r1, W4, b4, W5, b5, b2, rb);
  cnp_fused_dec<<<256, 256, 0, stream>>>(feats, posb, W1, W2, W5, rb, W6, b6, out);
}
